// Torch_Ops_Aten_QuantizedGru_InputModule_53987738910927
// MI455X (gfx1250) — matrix operand order repaired and checked by reading, compiled, not run on hardware
//
#include <hip/hip_runtime.h>
#include <hip/hip_bf16.h>

typedef __attribute__((ext_vector_type(16))) _Float16 v16h;
typedef __attribute__((ext_vector_type(8)))  _Float16 v8h;
typedef __attribute__((ext_vector_type(4)))  _Float16 v4h;
typedef __attribute__((ext_vector_type(8)))  float    v8f;
typedef __attribute__((ext_vector_type(4)))  unsigned int v4u;
typedef __attribute__((ext_vector_type(4)))  int      v4i;
typedef __attribute__((ext_vector_type(8)))  int      v8i;

namespace {
constexpr int   B_   = 32;
constexpr int   S_   = 512;
constexpr int   H_   = 512;
constexpr int   G3H  = 3 * H_;     // 1536
constexpr int   GM   = B_ * S_;    // 16384 GEMM rows
constexpr int   KD   = 1024;       // K for all layers (IN == 2H == 1024)
constexpr int   NWG  = 8;          // workgroups per direction in the scan
constexpr float ZPT  = 128.0f;
constexpr float SCL  = 1.0f / (127.0f * 32.0f);

// workspace layout (bytes)
constexpr size_t WIH_ELEMS = (size_t)3 * 2 * G3H * KD;        // 9,437,184 halves
constexpr size_t WHH_ELEMS = (size_t)3 * 2 * G3H * H_;        // 4,718,592 halves
constexpr size_t OFF_WIH  = 0;
constexpr size_t OFF_WHH  = OFF_WIH + WIH_ELEMS * 2;
constexpr size_t OFF_GI   = OFF_WHH + WHH_ELEMS * 2;          // 2 * GM * 1536 f32
constexpr size_t OFF_LB0  = OFF_GI  + (size_t)2 * GM * G3H * 4;
constexpr size_t OFF_LB1  = OFF_LB0 + (size_t)GM * 1024 * 4;
constexpr size_t OFF_HB   = OFF_LB1 + (size_t)GM * 1024 * 4;  // 2 dir x 2 buf x 32x512 f16
constexpr size_t OFF_CTR  = OFF_HB  + (size_t)2 * 2 * B_ * H_ * 2; // 2 dir x 512 ints
}

static __device__ __forceinline__ v16h cat8(v8h a, v8h b) {
  return __builtin_shufflevector(a, b, 0,1,2,3,4,5,6,7,8,9,10,11,12,13,14,15);
}
static __device__ __forceinline__ v8f wmma16(v16h a, v16h b, v8f c) {
  return __builtin_amdgcn_wmma_f32_16x16x32_f16(false, a, false, b, (short)0, c, false, false);
}

// ---------------------------------------------------------------------------
// TDM: load a 2D f16 tile (tile_d1 rows x tile_d0 halves, row stride in halves)
// from global to LDS, padding 8 halves after each 64-half row (pad_interval=4
// -> 32 DWORDs, pad_amount=3 -> 4 DWORDs) so the LDS row stride is 72 halves.
// ---------------------------------------------------------------------------
static __device__ __forceinline__ void tdm_load_b_tile(unsigned lds_off,
                                                       const _Float16* gptr,
                                                       int tile_d0, int tile_d1,
                                                       int stride_d0) {
  unsigned long long ga = (unsigned long long)(uintptr_t)gptr;
  v4u g0;
  g0[0] = 1u;                                          // count=1, user mode
  g0[1] = lds_off;                                     // lds_addr (bytes)
  g0[2] = (unsigned)(ga & 0xffffffffu);                // global_addr[31:0]
  g0[3] = (unsigned)((ga >> 32) & 0x1ffffffu)          // global_addr[56:32]
        | (2u << 30);                                  // type=2 ("image")
  v8i g1;
  g1[0] = (1 << 16)        // data_size = 1 -> 2 bytes
        | (1 << 20)        // pad_enable
        | (4 << 22)        // pad_interval: 32 DWORDs (one 64-half row)
        | (3 << 25);       // pad_amount: 4 DWORDs (8 halves)
  g1[1] = (tile_d0 << 16); // tensor_dim0[15:0] (tile fully in bounds)
  g1[2] = (tile_d1 << 16); // tensor_dim0 hi=0 | tensor_dim1[15:0]
  g1[3] = (tile_d0 << 16); // tensor_dim1 hi=0 | tile_dim0
  g1[4] = tile_d1;         // tile_dim1 | tile_dim2=0
  g1[5] = stride_d0;       // tensor_dim0_stride[31:0] (data_size units)
  g1[6] = 0;
  g1[7] = 0;
  v4i z4 = {0, 0, 0, 0};
#if defined(__clang_major__) && (__clang_major__ >= 23)
  v8i z8 = {0, 0, 0, 0, 0, 0, 0, 0};
  __builtin_amdgcn_tensor_load_to_lds(g0, g1, z4, z4, z8, 0);
#else
  __builtin_amdgcn_tensor_load_to_lds(g0, g1, z4, z4, 0);
#endif
}

// ---------------------------------------------------------------------------
// 1) dequantize int32 (0..255) -> f16 : (q - 128) * SCALE
// ---------------------------------------------------------------------------
__global__ __launch_bounds__(256)
void dequant_q_f16(const int* __restrict__ q, _Float16* __restrict__ o) {
  int i = blockIdx.x * 256 + threadIdx.x;
  int4 v = reinterpret_cast<const int4*>(q)[i];
  v4h r;
  r[0] = (_Float16)(((float)v.x - ZPT) * SCL);
  r[1] = (_Float16)(((float)v.y - ZPT) * SCL);
  r[2] = (_Float16)(((float)v.z - ZPT) * SCL);
  r[3] = (_Float16)(((float)v.w - ZPT) * SCL);
  reinterpret_cast<v4h*>(o)[i] = r;
}

// ---------------------------------------------------------------------------
// 2) per-layer scan init: h[buf0] = hx(layer), zero the step barrier counters
// ---------------------------------------------------------------------------
__global__ __launch_bounds__(256)
void scan_init(const float* __restrict__ Hx, _Float16* __restrict__ hb,
               int* __restrict__ ctr, int layer) {
  int tid = blockIdx.x * 256 + threadIdx.x;          // 0 .. 32767
  int d = tid >> 14;
  int r = tid & 16383;
  hb[(size_t)d * 2 * B_ * H_ + r] = (_Float16)Hx[(size_t)(2 * layer + d) * B_ * H_ + r];
  if (tid < 2 * S_) ctr[tid] = 0;
}

// ---------------------------------------------------------------------------
// 3) gi = X(f32, GM x 1024) @ W^T (f16, N=1536 x K=1024) + bias, f32 out
//    block tile 64x128, BK=64, 256 threads = 8 waves, each wave 16x64 of C.
//    B tile staged by the Tensor Data Mover (wave 0 issues, TENSORcnt wait);
//    A tile staged by VALU (f32 -> f16 conversion) overlapping the TDM DMA.
// ---------------------------------------------------------------------------
__global__ __launch_bounds__(256)
void gemm_gi(const float* __restrict__ X,
             const _Float16* __restrict__ Wl,   // (2, 1536, 1024) f16
             const float* __restrict__ biasl,   // (2, 1536)
             float* __restrict__ Gil) {         // (2, GM, 1536)
  constexpr int BM = 64, BN = 128, BK = 64, PAD = 8;
  const int dir = blockIdx.z;
  const _Float16* Wd  = Wl + (size_t)dir * G3H * KD;
  const float* bias   = biasl + dir * G3H;
  float* Gi           = Gil + (size_t)dir * GM * G3H;
  const int n0 = blockIdx.x * BN;
  const int m0 = blockIdx.y * BM;

  __shared__ _Float16 As[BM][BK + PAD];
  __shared__ _Float16 Bs[BN][BK + PAD];

  const int tid   = threadIdx.x;
  const int lane  = tid & 31;
  const int wave  = tid >> 5;
  const int mt    = wave & 3;    // 4 waves along M
  const int nh    = wave >> 2;   // 2 waves along N (64 each)
  const int lhalf = lane >> 4;
  const int l16   = lane & 15;

  const unsigned bs_lds = (unsigned)(uintptr_t)&Bs[0][0];

  v8f acc[4];
#pragma unroll
  for (int j = 0; j < 4; ++j)
#pragma unroll
    for (int r = 0; r < 8; ++r) acc[j][r] = 0.0f;

  for (int k0 = 0; k0 < KD; k0 += BK) {
    // ---- B tile: TDM async copy global->LDS (wave 0 only) ----
    if (tid < 32) {
      tdm_load_b_tile(bs_lds, Wd + (size_t)n0 * KD + k0, BK, BN, KD);
    }
    // ---- A tile: f32 -> f16 on VALU, overlaps the TDM transfer ----
    {
      int row = tid >> 2, kc = (tid & 3) * 16;
      const float* xp = X + (size_t)(m0 + row) * KD + k0 + kc;
      _Float16* ap = &As[row][kc];
#pragma unroll
      for (int u = 0; u < 2; ++u) {
        float4 f0 = *reinterpret_cast<const float4*>(xp + u * 8);
        float4 f1 = *reinterpret_cast<const float4*>(xp + u * 8 + 4);
        ap[u * 8 + 0] = (_Float16)f0.x; ap[u * 8 + 1] = (_Float16)f0.y;
        ap[u * 8 + 2] = (_Float16)f0.z; ap[u * 8 + 3] = (_Float16)f0.w;
        ap[u * 8 + 4] = (_Float16)f1.x; ap[u * 8 + 5] = (_Float16)f1.y;
        ap[u * 8 + 6] = (_Float16)f1.z; ap[u * 8 + 7] = (_Float16)f1.w;
      }
      if (k0 + BK < KD) __builtin_prefetch(xp + BK, 0, 3);
    }
    if (tid < 32) __builtin_amdgcn_s_wait_tensorcnt(0);
    __syncthreads();

    const int arow = mt * 16 + l16;
#pragma unroll
    for (int kc2 = 0; kc2 < 2; ++kc2) {
      v8h alo = *reinterpret_cast<const v8h*>(&As[arow][kc2 * 32 + lhalf * 8]);
      v8h ahi = *reinterpret_cast<const v8h*>(&As[arow][kc2 * 32 + 16 + lhalf * 8]);
      v16h afr = cat8(alo, ahi);
#pragma unroll
      for (int j = 0; j < 4; ++j) {
        const int brow = nh * 64 + j * 16 + l16;
        v8h blo = *reinterpret_cast<const v8h*>(&Bs[brow][kc2 * 32 + lhalf * 8]);
        v8h bhi = *reinterpret_cast<const v8h*>(&Bs[brow][kc2 * 32 + 16 + lhalf * 8]);
        acc[j] = wmma16(afr, cat8(blo, bhi), acc[j]);
      }
    }
    __syncthreads();
  }

#pragma unroll
  for (int j = 0; j < 4; ++j) {
    const int n = n0 + nh * 64 + j * 16 + l16;
    const float bv = bias[n];
#pragma unroll
    for (int r = 0; r < 8; ++r) {
      const int m = m0 + mt * 16 + lhalf * 8 + r;
      Gi[(size_t)m * G3H + n] = acc[j][r] + bv;
    }
  }
}

// ---------------------------------------------------------------------------
// 4) multi-workgroup GRU scan. grid = (NWG, 2 dirs), 256 threads (8 waves).
//    WG g owns hidden columns [g*64, g*64+64) and the 3 matching gate strips
//    (12 n-tiles of gh). h is double-buffered f16 in global (L2-resident);
//    per-step device barrier via a per-step atomic counter.
// ---------------------------------------------------------------------------
__global__ __launch_bounds__(256)
void gru_scan(const float* __restrict__ GiAll,     // (2, GM, 1536)
              const _Float16* __restrict__ WhhL,   // (2, 1536, 512) f16
              const float* __restrict__ BhhL,      // (2, 1536)
              float* __restrict__ Ys,              // (32, 512, 1024) concat
              float* __restrict__ HfL,             // hfin + 2*layer*B*H
              _Float16* __restrict__ hb_all,       // (2 dir, 2 buf, 32, 512) f16
              int* __restrict__ ctr_all) {         // (2 dir, 512)
  const int g = blockIdx.x;            // 0..NWG-1 : hidden-column slice
  const int d = blockIdx.y;            // direction
  const float*    Gi  = GiAll + (size_t)d * GM * G3H;
  const _Float16* Whh = WhhL + (size_t)d * G3H * H_;
  const float*    Bhh = BhhL + d * G3H;
  float*          Hf  = HfL + (size_t)d * B_ * H_;
  _Float16*       hb  = hb_all + (size_t)d * 2 * B_ * H_;
  int*            ctr = ctr_all + d * S_;

  __shared__ float ghb[B_][200];       // 12 tiles * 16 cols (+8 pad)

  const int tid   = threadIdx.x;
  const int lane  = tid & 31;
  const int wave  = tid >> 5;          // 0..7
  const int mt    = wave & 1;          // 2 waves along M (32 rows)
  const int tbase = (wave >> 1) * 3;   // 3 local n-tiles per wave (12 total)
  const int lhalf = lane >> 4;
  const int l16   = lane & 15;
  const int arow  = mt * 16 + l16;

  for (int t = 0; t < S_; ++t) {
    const _Float16* hcur = hb + (size_t)(t & 1) * B_ * H_;
    _Float16*       hnxt = hb + (size_t)((t + 1) & 1) * B_ * H_;

    // ---- gh slice = h @ whh^T for this WG's 12 n-tiles ----
    v8f acc[3];
#pragma unroll
    for (int j = 0; j < 3; ++j)
#pragma unroll
      for (int r = 0; r < 8; ++r) acc[j][r] = 0.0f;

    for (int kc = 0; kc < H_ / 32; ++kc) {
      const _Float16* ap = hcur + (size_t)arow * H_ + kc * 32 + lhalf * 8;
      v8h alo = *reinterpret_cast<const v8h*>(ap);
      v8h ahi = *reinterpret_cast<const v8h*>(ap + 16);
      v16h afr = cat8(alo, ahi);
#pragma unroll
      for (int j = 0; j < 3; ++j) {
        const int tloc = tbase + j;                           // 0..11
        const int ntile = (tloc >> 2) * 32 + g * 4 + (tloc & 3);
        const _Float16* wp =
            Whh + (size_t)(ntile * 16 + l16) * H_ + kc * 32 + lhalf * 8;
        v8h blo = *reinterpret_cast<const v8h*>(wp);
        v8h bhi = *reinterpret_cast<const v8h*>(wp + 16);
        acc[j] = wmma16(afr, cat8(blo, bhi), acc[j]);
      }
    }
#pragma unroll
    for (int j = 0; j < 3; ++j) {
      const int c = (tbase + j) * 16 + l16;                   // local gh col
#pragma unroll
      for (int r = 0; r < 8; ++r)
        ghb[mt * 16 + lhalf * 8 + r][c] = acc[j][r];
    }
    __syncthreads();

    // ---- gates + h update for columns [g*64, g*64+64) ----
    const int s_idx = d ? (S_ - 1 - t) : t;
#pragma unroll
    for (int i = 0; i < 8; ++i) {
      int idx = tid + i * 256;                 // 0..2047
      int b = idx >> 6, jl = idx & 63;
      int jg = g * 64 + jl;
      const size_t gb = ((size_t)b * S_ + s_idx) * G3H;
      float ir  = Gi[gb + jg];
      float iz  = Gi[gb + H_ + jg];
      float in_ = Gi[gb + 2 * H_ + jg];
      float hr = ghb[b][jl]       + Bhh[jg];
      float hz = ghb[b][64 + jl]  + Bhh[H_ + jg];
      float hn = ghb[b][128 + jl] + Bhh[2 * H_ + jg];
      float rg = 1.0f / (1.0f + __expf(-(ir + hr)));
      float zg = 1.0f / (1.0f + __expf(-(iz + hz)));
      float ng = tanhf(in_ + rg * hn);
      float hold = (float)hcur[b * H_ + jg];
      float hnew = (1.0f - zg) * ng + zg * hold;
      Ys[((size_t)b * S_ + s_idx) * 1024 + (size_t)d * H_ + jg] = hnew;
      hnxt[b * H_ + jg] = (_Float16)hnew;
    }

    // ---- device-scope step barrier (per-step counter, no reset needed) ----
    if (t < S_ - 1) {
      if (tid == 0) {
        __threadfence();
        atomicAdd(&ctr[t], 1);
        while (__hip_atomic_load(&ctr[t], __ATOMIC_ACQUIRE,
                                 __HIP_MEMORY_SCOPE_AGENT) < NWG)
          __builtin_amdgcn_s_sleep(2);
      }
      __syncthreads();
    }
  }

  // final hidden state: h after step S-1 lives in buffer (S_ & 1) == 0
  const _Float16* hfinal = hb;   // our slice was written by this WG
#pragma unroll
  for (int i = 0; i < 8; ++i) {
    int idx = tid + i * 256;
    int b = idx >> 6, jl = idx & 63, jg = blockIdx.x * 64 + jl;
    Hf[b * H_ + jg] = (float)hfinal[b * H_ + jg];
  }
}

// ---------------------------------------------------------------------------
extern "C" void kernel_launch(void* const* d_in, const int* in_sizes, int n_in,
                              void* d_out, int out_size, void* d_ws, size_t ws_size,
                              hipStream_t stream) {
  (void)in_sizes; (void)n_in; (void)out_size; (void)ws_size;
  const float* x0   = (const float*)d_in[0];
  const float* hx   = (const float*)d_in[1];
  const int*   wihq = (const int*)d_in[2];
  const int*   whhq = (const int*)d_in[3];
  const float* bih  = (const float*)d_in[4];
  const float* bhh  = (const float*)d_in[5];
  float* out = (float*)d_out;

  char* ws = (char*)d_ws;
  _Float16* wih16 = (_Float16*)(ws + OFF_WIH);
  _Float16* whh16 = (_Float16*)(ws + OFF_WHH);
  float* gi    = (float*)(ws + OFF_GI);
  float* lbuf0 = (float*)(ws + OFF_LB0);
  float* lbuf1 = (float*)(ws + OFF_LB1);
  _Float16* hbufs = (_Float16*)(ws + OFF_HB);
  int* ctr = (int*)(ws + OFF_CTR);

  // dequantize weights
  dequant_q_f16<<<(int)(WIH_ELEMS / 4 / 256), 256, 0, stream>>>(wihq, wih16);
  dequant_q_f16<<<(int)(WHH_ELEMS / 4 / 256), 256, 0, stream>>>(whhq, whh16);

  float* hfin = out + (size_t)B_ * S_ * 2 * H_;   // second output section

  for (int l = 0; l < 3; ++l) {
    const float* X = (l == 0) ? x0 : ((l == 1) ? lbuf0 : lbuf1);
    float* Y = (l == 2) ? out : ((l == 0) ? lbuf0 : lbuf1);

    gemm_gi<<<dim3(G3H / 128, GM / 64, 2), 256, 0, stream>>>(
        X, wih16 + (size_t)l * 2 * G3H * KD, bih + (size_t)l * 2 * G3H, gi);

    scan_init<<<128, 256, 0, stream>>>(hx, hbufs, ctr, l);

    gru_scan<<<dim3(NWG, 2), 256, 0, stream>>>(
        gi, whh16 + (size_t)l * 2 * G3H * H_, bhh + (size_t)l * 2 * G3H,
        Y, hfin + (size_t)2 * l * B_ * H_, hbufs, ctr);
  }
}
